// Memory_30090540875982
// MI455X (gfx1250) — matrix operand order repaired and checked by reading, not compiled, not run on hardware
//
#include <hip/hip_runtime.h>

typedef __attribute__((ext_vector_type(16))) _Float16 v16h;
typedef __attribute__((ext_vector_type(8)))  _Float16 v8h;
typedef __attribute__((ext_vector_type(8)))  float    v8f;

#define NB    4
#define DE    128
#define DO    512
#define NMEM  12960          // 8*30*54  (softmax axis)
#define NQ    1620           // 30*54
#define NQT   102            // ceil(1620/16)
#define SCALE 0.08838834764831845f   // 1/sqrt(128)

// ---------------------------------------------------------------------------
// K0a: transpose m_in [B,DE,NMEM] f32  ->  A1 [B,NMEM,DE] f16  (LDS tiled)
// ---------------------------------------------------------------------------
__global__ void k_transpose_min(const float* __restrict__ in, _Float16* __restrict__ out) {
  __shared__ float tile[32][33];
  const int b = blockIdx.z, d0 = blockIdx.y * 32, m0 = blockIdx.x * 32;
  const int tx = threadIdx.x, ty = threadIdx.y;
#pragma unroll
  for (int i = 0; i < 4; ++i) {
    int d = d0 + ty + i * 8;
    tile[ty + i * 8][tx] = in[((size_t)b * DE + d) * NMEM + m0 + tx];
  }
  __syncthreads();
#pragma unroll
  for (int i = 0; i < 4; ++i) {
    int m = m0 + ty + i * 8;
    out[((size_t)b * NMEM + m) * DE + d0 + tx] = (_Float16)tile[tx][ty + i * 8];
  }
}

// ---------------------------------------------------------------------------
// K0b: elementwise convert m_out f32 -> f16 (same layout [B,DO,NMEM])
// ---------------------------------------------------------------------------
__global__ void k_cvt_f16(const float* __restrict__ in, _Float16* __restrict__ out, size_t n) {
  size_t i = (size_t)blockIdx.x * blockDim.x + threadIdx.x;
  size_t stride = (size_t)gridDim.x * blockDim.x;
  for (; i < n; i += stride) out[i] = (_Float16)in[i];
}

// ---------------------------------------------------------------------------
// K1: logits  p[b,m,q] = (A1[b,m,:] . q_in[b,:,q]) * SCALE
//     one wave -> 32(m) x 16(q) tile, K=128 in 4 WMMA steps, 2 acc frags
// ---------------------------------------------------------------------------
__global__ void k_gemm1(const _Float16* __restrict__ A1, const float* __restrict__ qin,
                        float* __restrict__ p) {
  const int wid  = threadIdx.x >> 5;
  const int lane = threadIdx.x & 31;
  const int qt = blockIdx.x * 4 + wid;
  if (qt >= NQT) return;                       // uniform per wave
  const int b  = blockIdx.z;
  const int m0 = blockIdx.y * 32;
  const int l15 = lane & 15, half = lane >> 4;
  const int q  = qt * 16 + l15;
  const int qc = q < NQ ? q : NQ - 1;

  v8f acc0 = {}, acc1 = {};
  const float*    qbase = qin + (size_t)b * DE * NQ + qc;
  const _Float16* abase = A1 + ((size_t)b * NMEM + m0 + l15) * DE + half * 8;

#pragma unroll
  for (int k0 = 0; k0 < DE; k0 += 32) {
    // B frag: lane = column q, K rows 8h..8h+7 then 16+8h..16+8h+7 (h = lane>=16)
    v16h bfr;
    const int klo = k0 + half * 8;
#pragma unroll
    for (int kk = 0; kk < 16; ++kk)
      bfr[kk] = (_Float16)qbase[(size_t)(klo + kk + (kk & 8)) * NQ];

    // A frags: row m0+l15 (+16 for second frag), K chunks kb and kb+16
    v16h a0, a1;
    {
      const _Float16* ap = abase + k0;
      v8h lo = *(const v8h*)ap;
      v8h hi = *(const v8h*)(ap + 16);
#pragma unroll
      for (int i = 0; i < 8; ++i) { a0[i] = lo[i]; a0[8 + i] = hi[i]; }
      ap += (size_t)16 * DE;
      lo = *(const v8h*)ap;
      hi = *(const v8h*)(ap + 16);
#pragma unroll
      for (int i = 0; i < 8; ++i) { a1[i] = lo[i]; a1[8 + i] = hi[i]; }
    }
    acc0 = __builtin_amdgcn_wmma_f32_16x16x32_f16(false, a0, false, bfr, (short)0, acc0, false, false);
    acc1 = __builtin_amdgcn_wmma_f32_16x16x32_f16(false, a1, false, bfr, (short)0, acc1, false, false);
  }

  if (q < NQ) {
    float* pb = p + ((size_t)b * NMEM + m0 + half * 8) * NQ + q;
#pragma unroll
    for (int r = 0; r < 8; ++r) pb[(size_t)r * NQ] = acc0[r] * SCALE;
    pb += (size_t)16 * NQ;
#pragma unroll
    for (int r = 0; r < 8; ++r) pb[(size_t)r * NQ] = acc1[r] * SCALE;
  }
}

// ---------------------------------------------------------------------------
// K2: softmax over m (12960) per (b,q) column, in place on p.
//     block = (64 q, 4 m-strides): every row access is a 256B coalesced burst.
//     Pass 1: online max+sum; LDS combine; Pass 2: normalize.
// ---------------------------------------------------------------------------
__global__ void k_softmax(float* __restrict__ p) {
  const int b  = blockIdx.y;
  const int tx = threadIdx.x, ty = threadIdx.y;
  const int q  = blockIdx.x * 64 + tx;
  __shared__ float smax[4][64];
  __shared__ float ssum[4][64];

  float M = -3.4e38f, S = 0.f;
  float* col = p + (size_t)b * NMEM * NQ + q;
  if (q < NQ) {
    for (int m = ty; m < NMEM; m += 4) {
      float x  = col[(size_t)m * NQ];
      float nm = fmaxf(M, x);
      S = S * __expf(M - nm) + __expf(x - nm);
      M = nm;
    }
  }
  smax[ty][tx] = M; ssum[ty][tx] = S;
  __syncthreads();
  if (ty == 0) {
    float Mm = smax[0][tx], Ss = ssum[0][tx];
#pragma unroll
    for (int j = 1; j < 4; ++j) {
      float nm = fmaxf(Mm, smax[j][tx]);
      Ss = Ss * __expf(Mm - nm) + ssum[j][tx] * __expf(smax[j][tx] - nm);
      Mm = nm;
    }
    smax[0][tx] = Mm; ssum[0][tx] = 1.0f / Ss;
  }
  __syncthreads();
  const float Mf = smax[0][tx], R = ssum[0][tx];
  if (q < NQ) {
    for (int m = ty; m < NMEM; m += 4)
      col[(size_t)m * NQ] = __expf(col[(size_t)m * NQ] - Mf) * R;
  }
}

// ---------------------------------------------------------------------------
// K3: mem[b,c,q] = sum_m A2[b,c,m] * p[b,m,q]
//     one wave -> 64(c) x 16(q) tile; B frag (p column block) shared by 4 accs
// ---------------------------------------------------------------------------
__global__ void k_gemm2(const _Float16* __restrict__ A2, const float* __restrict__ p,
                        float* __restrict__ out) {
  const int wid  = threadIdx.x >> 5;
  const int lane = threadIdx.x & 31;
  const int qt = blockIdx.x * 4 + wid;
  if (qt >= NQT) return;                       // uniform per wave
  const int b  = blockIdx.z;
  const int c0 = blockIdx.y * 64;
  const int l15 = lane & 15, half = lane >> 4;
  const int q  = qt * 16 + l15;
  const int qc = q < NQ ? q : NQ - 1;

  v8f acc[4] = {{}, {}, {}, {}};
  const float*    pbase = p  + (size_t)b * NMEM * NQ + qc;
  const _Float16* abase = A2 + ((size_t)b * DO + c0 + l15) * NMEM + half * 8;

  for (int m0 = 0; m0 < NMEM; m0 += 32) {
    v16h bfr;
    const float* bp = pbase + (size_t)(m0 + half * 8) * NQ;
#pragma unroll
    for (int kk = 0; kk < 16; ++kk)
      bfr[kk] = (_Float16)bp[(size_t)(kk + (kk & 8)) * NQ];

#pragma unroll
    for (int ci = 0; ci < 4; ++ci) {
      const _Float16* ap = abase + (size_t)(ci * 16) * NMEM + m0;
      v8h lo = *(const v8h*)ap;
      v8h hi = *(const v8h*)(ap + 16);
      v16h a;
#pragma unroll
      for (int i = 0; i < 8; ++i) { a[i] = lo[i]; a[8 + i] = hi[i]; }
      acc[ci] = __builtin_amdgcn_wmma_f32_16x16x32_f16(false, a, false, bfr, (short)0, acc[ci], false, false);
    }
  }

  if (q < NQ) {
#pragma unroll
    for (int ci = 0; ci < 4; ++ci) {
      float* ob = out + ((size_t)b * (2 * DO) + c0 + ci * 16 + half * 8) * NQ + q;
#pragma unroll
      for (int r = 0; r < 8; ++r) ob[(size_t)r * NQ] = acc[ci][r];
    }
  }
}

// ---------------------------------------------------------------------------
// K4: copy q_out [B,DO,NQ] into channels [DO..2*DO) of the output
// ---------------------------------------------------------------------------
__global__ void k_copy_qout(const float* __restrict__ qout, float* __restrict__ out) {
  size_t i = (size_t)blockIdx.x * blockDim.x + threadIdx.x;
  const size_t n = (size_t)NB * DO * NQ;
  const size_t stride = (size_t)gridDim.x * blockDim.x;
  const size_t per_b = (size_t)DO * NQ;
  for (; i < n; i += stride) {
    size_t b = i / per_b, rem = i % per_b;
    out[(b * (2 * DO) + DO) * (size_t)NQ + rem] = qout[i];
  }
}

// ---------------------------------------------------------------------------
extern "C" void kernel_launch(void* const* d_in, const int* in_sizes, int n_in,
                              void* d_out, int out_size, void* d_ws, size_t ws_size,
                              hipStream_t stream) {
  (void)in_sizes; (void)n_in; (void)out_size; (void)ws_size;
  const float* m_in  = (const float*)d_in[0];   // [B,128,8,30,54]
  const float* m_out = (const float*)d_in[1];   // [B,512,8,30,54]
  const float* q_in  = (const float*)d_in[2];   // [B,128,30,54]
  const float* q_out = (const float*)d_in[3];   // [B,512,30,54]

  float* out = (float*)d_out;                       // mem_out: B*1024*1620 floats
  float* p   = out + (size_t)NB * (2 * DO) * NQ;    // p: B*12960*1620 floats

  _Float16* A1 = (_Float16*)d_ws;                   // [B,NMEM,DE]  (13.3 MB)
  _Float16* A2 = A1 + (size_t)NB * NMEM * DE;       // [B,DO,NMEM]  (53.1 MB)

  k_transpose_min<<<dim3(NMEM / 32, DE / 32, NB), dim3(32, 8), 0, stream>>>(m_in, A1);
  k_cvt_f16      <<<dim3(2048), dim3(256), 0, stream>>>(m_out, A2, (size_t)NB * DO * NMEM);
  k_gemm1        <<<dim3((NQT + 3) / 4, NMEM / 32, NB), dim3(128), 0, stream>>>(A1, q_in, p);
  k_softmax      <<<dim3((NQ + 63) / 64, NB), dim3(64, 4), 0, stream>>>(p);
  k_gemm2        <<<dim3((NQT + 3) / 4, DO / 64, NB), dim3(128), 0, stream>>>(A2, p, out);
  k_copy_qout    <<<dim3(2048), dim3(256), 0, stream>>>(q_out, out);
}
